// Block_50818053046338
// MI455X (gfx1250) — compile-verified, matrix operand order checked by reading, not run on hardware
//
#include <hip/hip_runtime.h>
#include <hip/hip_bf16.h>
#include <math.h>

// ---------------- problem constants ----------------
#define B_   16
#define C_   768
#define P_   2048
#define NP_  64
#define C4_  3072

typedef __attribute__((ext_vector_type(16))) __bf16 v16bf;
typedef __attribute__((ext_vector_type(8)))  __bf16 v8bf;
typedef __attribute__((ext_vector_type(8)))  float  v8f;

__device__ __forceinline__ float gelu_erf(float x) {
  return 0.5f * x * (1.0f + erff(x * 0.70710678118654752440f));
}

// CDNA5 async global->LDS copy, 16B per lane, tracked by ASYNCcnt.
// LDS address = low 32 bits of the generic pointer.
__device__ __forceinline__ void async_g2l_b128(void* lds_ptr, const void* gptr) {
  unsigned lds_addr = (unsigned)(unsigned long long)lds_ptr;
  asm volatile("global_load_async_to_lds_b128 %0, %1, off"
               :: "v"(lds_addr), "v"(gptr)
               : "memory");
}
__device__ __forceinline__ void wait_asynccnt0() {
  asm volatile("s_wait_asynccnt 0" ::: "memory");
}

// ---------------- small utility kernels ----------------

// fp32 [K,N] -> bf16 [N,K] (convert + transpose; for weights, done once per launch)
__global__ __launch_bounds__(256) void cvtT_kernel(const float* __restrict__ in,
                                                   __bf16* __restrict__ out,
                                                   int K, int N) {
  __shared__ float tile[32][33];
  int n0 = blockIdx.x * 32;
  int k0 = blockIdx.y * 32;
  int tx = threadIdx.x, ty = threadIdx.y;
  #pragma unroll
  for (int j = 0; j < 4; ++j)
    tile[ty + j * 8][tx] = in[(long)(k0 + ty + j * 8) * N + n0 + tx];
  __syncthreads();
  #pragma unroll
  for (int j = 0; j < 4; ++j)
    out[(long)(n0 + ty + j * 8) * K + k0 + tx] = (__bf16)tile[tx][ty + j * 8];
}

// x[B,C,P] -> xt[B,P,C], 32x32 LDS tiles
__global__ __launch_bounds__(256) void transpose_kernel(const float* __restrict__ x,
                                                        float* __restrict__ xt) {
  __shared__ float tile[32][33];
  int b  = blockIdx.z;
  int p0 = blockIdx.x * 32;
  int c0 = blockIdx.y * 32;
  int tx = threadIdx.x, ty = threadIdx.y;
  const float* xb = x + (long)b * C_ * P_;
  #pragma unroll
  for (int j = 0; j < 4; ++j)
    tile[ty + j * 8][tx] = xb[(long)(c0 + ty + j * 8) * P_ + p0 + tx];
  __syncthreads();
  float* xtb = xt + (long)b * P_ * C_;
  #pragma unroll
  for (int j = 0; j < 4; ++j)
    xtb[(long)(p0 + ty + j * 8) * C_ + c0 + tx] = tile[tx][ty + j * 8];
}

// LayerNorm over rows of length 768; writes up to 3 affine bf16 outputs.
__global__ __launch_bounds__(256) void ln3_kernel(
    const float* __restrict__ in,
    __bf16* __restrict__ o0, const float* __restrict__ g0, const float* __restrict__ b0,
    __bf16* __restrict__ o1, const float* __restrict__ g1, const float* __restrict__ b1,
    __bf16* __restrict__ o2, const float* __restrict__ g2, const float* __restrict__ b2) {
  long row = blockIdx.x;
  const float* x = in + row * (long)C_;
  int t = threadIdx.x;
  float v0 = x[t], v1 = x[t + 256], v2 = x[t + 512];
  float s  = v0 + v1 + v2;
  float ss = v0 * v0 + v1 * v1 + v2 * v2;
  __shared__ float rs[256], rss[256];
  rs[t] = s; rss[t] = ss;
  __syncthreads();
  for (int off = 128; off > 0; off >>= 1) {
    if (t < off) { rs[t] += rs[t + off]; rss[t] += rss[t + off]; }
    __syncthreads();
  }
  float mean = rs[0] * (1.0f / 768.0f);
  float var  = rss[0] * (1.0f / 768.0f) - mean * mean;
  float rstd = rsqrtf(var + 1e-5f);
  float vv[3] = {v0, v1, v2};
  #pragma unroll
  for (int j = 0; j < 3; ++j) {
    int c = t + j * 256;
    float xh = (vv[j] - mean) * rstd;
    if (o0) o0[row * C_ + c] = (__bf16)(xh * g0[c] + b0[c]);
    if (o1) o1[row * C_ + c] = (__bf16)(xh * g1[c] + b1[c]);
    if (o2) o2[row * C_ + c] = (__bf16)(xh * g2[c] + b2[c]);
  }
}

// SimpleReasoning gate: parts[B,NP,C] -> parts1 = (sigmoid(gelu(max_c @w1+b1)@w2+b2)+1)*parts
__global__ __launch_bounds__(64) void sr_gate_kernel(
    const float* __restrict__ parts,
    const float* __restrict__ w1, const float* __restrict__ b1,
    const float* __restrict__ w2, const float* __restrict__ b2,
    float* __restrict__ parts1) {
  int b = blockIdx.x, t = threadIdx.x;
  const float* pb = parts + (long)b * NP_ * C_;
  float mx = -1e30f;
  for (int c = 0; c < C_; ++c) mx = fmaxf(mx, pb[(long)t * C_ + c]);
  __shared__ float sg[NP_], sh[NP_], sgate[NP_];
  sg[t] = mx;
  __syncthreads();
  float a = b1[t];
  for (int i = 0; i < NP_; ++i) a += sg[i] * w1[i * NP_ + t];
  a = gelu_erf(a);
  sh[t] = a;
  __syncthreads();
  float g = b2[t];
  for (int i = 0; i < NP_; ++i) g += sh[i] * w2[i * NP_ + t];
  sgate[t] = 1.0f / (1.0f + expf(-g)) + 1.0f;
  __syncthreads();
  float* ob = parts1 + (long)b * NP_ * C_;
  for (int idx = t; idx < NP_ * C_; idx += 64) {
    int r = idx / C_;
    ob[idx] = sgate[r] * pb[idx];
  }
}

// enc softmax: rows = B*NP, width P. S already relu*scale. rowmax -> attn_0.
__global__ __launch_bounds__(256) void softmax_enc_kernel(
    const float* __restrict__ S, __bf16* __restrict__ Pr, float* __restrict__ attn0) {
  long row = blockIdx.x;
  const float* s = S + row * (long)P_;
  int t = threadIdx.x;
  float v[8];
  float mx = -1e30f;
  #pragma unroll
  for (int j = 0; j < 8; ++j) { v[j] = s[t + j * 256]; mx = fmaxf(mx, v[j]); }
  __shared__ float red[256];
  red[t] = mx;
  __syncthreads();
  for (int off = 128; off > 0; off >>= 1) {
    if (t < off) red[t] = fmaxf(red[t], red[t + off]);
    __syncthreads();
  }
  float rmax = red[0];
  __syncthreads();
  if (t == 0) attn0[row] = rmax;
  float sum = 0.0f;
  #pragma unroll
  for (int j = 0; j < 8; ++j) { v[j] = expf(v[j] - rmax); sum += v[j]; }
  red[t] = sum;
  __syncthreads();
  for (int off = 128; off > 0; off >>= 1) {
    if (t < off) red[t] += red[t + off];
    __syncthreads();
  }
  float inv = 1.0f / red[0];
  #pragma unroll
  for (int j = 0; j < 8; ++j) Pr[row * (long)P_ + t + j * 256] = (__bf16)(v[j] * inv);
}

// dec softmax: rows = B*P, width NP=64. S already relu*scale.
__global__ __launch_bounds__(64) void softmax_dec_kernel(
    const float* __restrict__ S, __bf16* __restrict__ Pr) {
  long row = blockIdx.x;
  const float* s = S + row * (long)NP_;
  int t = threadIdx.x;
  float v = s[t];
  __shared__ float red[64];
  red[t] = v;
  __syncthreads();
  for (int off = 32; off > 0; off >>= 1) {
    if (t < off) red[t] = fmaxf(red[t], red[t + off]);
    __syncthreads();
  }
  float rmax = red[0];
  __syncthreads();
  float e = expf(v - rmax);
  red[t] = e;
  __syncthreads();
  for (int off = 32; off > 0; off >>= 1) {
    if (t < off) red[t] += red[t + off];
    __syncthreads();
  }
  Pr[row * (long)NP_ + t] = (__bf16)(e / red[0]);
}

// ---------------- WMMA GEMM ----------------
// Fragment per CDNA5 ISA bf16 A layout: lane = (m|n)%16, half = lane>>4 picks K-halves.
__device__ __forceinline__ v16bf frag_from_lds(const __bf16* rowptr, int half) {
  v8bf lo = *(const v8bf*)(rowptr + 8 * half);
  v8bf hi = *(const v8bf*)(rowptr + 16 + 8 * half);
  return __builtin_shufflevector(lo, hi, 0, 1, 2, 3, 4, 5, 6, 7,
                                 8, 9, 10, 11, 12, 13, 14, 15);
}

// D = act(A @ B^T + bias) (+resid).  A is [M,K] row-major, B is [N,K] row-major.
// M % (MW*32) == 0, N % (NW*64) == 0, K % 32 == 0 are REQUIRED (no guards).
// ACT: 0 none, 1 gelu, 2 relu*scale.
// OUTM: 0 bf16 [*,ldc]; 1 fp32 [*,ldc]; 2 fp32 transposed (gm=b*ldc+p -> [b][gn][p], batch stride sCz);
//       3 bf16 transposed (same indexing as 2; requires ldc % BM == 0 or BM % ldc-tile fit).
template <int MW, int NW, int ACT, int OUTM, bool RES>
__global__ __launch_bounds__(256) void gemm_wmma_kernel(
    const __bf16* __restrict__ A, long sAz, int lda,
    const __bf16* __restrict__ Bm, long sBz, int ldb,
    const float* __restrict__ bias,
    const float* __restrict__ resid, long sRz, int ldr,
    void* __restrict__ Cout, long sCz, int ldc,
    int K, float scale) {
  constexpr int BM   = MW * 32;
  constexpr int BN   = NW * 64;
  constexpr int LDK  = 40;            // 32 + pad, rows are 80B (16B aligned)
  constexpr int ASEG = (BM * 4) / 256;  // 16B segments per thread for A tile
  constexpr int BSEG = (BN * 4) / 256;
  __shared__ __bf16 sA[2][BM * LDK];
  __shared__ __bf16 sB[2][BN * LDK];

  const int tid  = threadIdx.x;
  const int lane = tid & 31;
  const int l15  = lane & 15;
  const int half = lane >> 4;
  const int wave = tid >> 5;
  const int wm   = wave % MW;
  const int wn   = wave / MW;
  const int z    = blockIdx.z;
  const int m0   = blockIdx.y * BM;
  const int n0   = blockIdx.x * BN;

  A  += (long)z * sAz + (long)m0 * lda;
  Bm += (long)z * sBz + (long)n0 * ldb;
  const float* rz = resid;
  if (RES) rz += (long)z * sRz;

  // issue one K-step worth of async global->LDS copies into buffer `buf`
  auto stage = [&](int buf, int k0) {
    #pragma unroll
    for (int i = 0; i < ASEG; ++i) {
      int idx = tid + i * 256;
      int r = idx >> 2, cg = (idx & 3) << 3;
      async_g2l_b128(&sA[buf][r * LDK + cg], A + (long)r * lda + (k0 + cg));
    }
    #pragma unroll
    for (int i = 0; i < BSEG; ++i) {
      int idx = tid + i * 256;
      int r = idx >> 2, cg = (idx & 3) << 3;
      async_g2l_b128(&sB[buf][r * LDK + cg], Bm + (long)r * ldb + (k0 + cg));
    }
  };

  v8f acc[2][4];
  #pragma unroll
  for (int i = 0; i < 2; ++i)
    #pragma unroll
    for (int j = 0; j < 4; ++j)
      #pragma unroll
      for (int e = 0; e < 8; ++e) acc[i][j][e] = 0.0f;

  stage(0, 0);
  int buf = 0;
  for (int k0 = 0; k0 < K; k0 += 32) {
    wait_asynccnt0();    // my async stores into `buf` have landed in LDS
    __syncthreads();     // everyone's have; also: all reads of `buf^1` finished
    if (k0 + 32 < K) stage(buf ^ 1, k0 + 32);

    v16bf af[2];
    #pragma unroll
    for (int mt = 0; mt < 2; ++mt)
      af[mt] = frag_from_lds(&sA[buf][(wm * 32 + mt * 16 + l15) * LDK], half);
    v16bf bfr[4];
    #pragma unroll
    for (int nt = 0; nt < 4; ++nt)
      bfr[nt] = frag_from_lds(&sB[buf][(wn * 64 + nt * 16 + l15) * LDK], half);

    #pragma unroll
    for (int mt = 0; mt < 2; ++mt)
      #pragma unroll
      for (int nt = 0; nt < 4; ++nt)
        acc[mt][nt] = __builtin_amdgcn_wmma_f32_16x16x32_bf16(
            false, af[mt], false, bfr[nt], (short)0, acc[mt][nt], false, false);
    buf ^= 1;
  }

  // ---- epilogue ----
  int bb = 0, pbase = 0;
  if constexpr (OUTM >= 2) { bb = m0 / ldc; pbase = m0 - bb * ldc; }
  #pragma unroll
  for (int nt = 0; nt < 4; ++nt) {
    int gn = n0 + wn * 64 + nt * 16 + l15;
    float bv = bias ? bias[gn] : 0.0f;
    #pragma unroll
    for (int mt = 0; mt < 2; ++mt) {
      #pragma unroll
      for (int r = 0; r < 8; ++r) {
        int mo = wm * 32 + mt * 16 + half * 8 + r;  // row within block tile
        int gm = m0 + mo;
        float v = acc[mt][nt][r] + bv;
        if constexpr (ACT == 1) v = gelu_erf(v);
        if constexpr (ACT == 2) v = fmaxf(v, 0.0f) * scale;
        if constexpr (RES) v += rz[(long)gm * ldr + gn];
        if constexpr (OUTM == 0) {
          ((__bf16*)Cout)[(long)z * sCz + (long)gm * ldc + gn] = (__bf16)v;
        } else if constexpr (OUTM == 1) {
          ((float*)Cout)[(long)z * sCz + (long)gm * ldc + gn] = v;
        } else if constexpr (OUTM == 2) {
          ((float*)Cout)[(long)bb * sCz + (long)gn * ldc + (pbase + mo)] = v;
        } else {
          ((__bf16*)Cout)[(long)bb * sCz + (long)gn * ldc + (pbase + mo)] = (__bf16)v;
        }
      }
    }
  }
}

// ---------------- host launcher ----------------
extern "C" void kernel_launch(void* const* d_in, const int* in_sizes, int n_in,
                              void* d_out, int out_size, void* d_ws, size_t ws_size,
                              hipStream_t stream) {
  (void)in_sizes; (void)n_in; (void)out_size; (void)ws_size;

  const float* x     = (const float*)d_in[0];
  const float* parts = (const float*)d_in[1];
  const float* enc_nq_g = (const float*)d_in[2];
  const float* enc_nq_b = (const float*)d_in[3];
  const float* enc_nk_g = (const float*)d_in[4];
  const float* enc_nk_b = (const float*)d_in[5];
  const float* enc_nv_g = (const float*)d_in[6];
  const float* enc_nv_b = (const float*)d_in[7];
  const float* enc_wq = (const float*)d_in[8];
  const float* enc_wk = (const float*)d_in[9];
  const float* enc_wv = (const float*)d_in[10];
  const float* enc_wp = (const float*)d_in[11];
  const float* enc_bq = (const float*)d_in[12];
  const float* enc_bk = (const float*)d_in[13];
  const float* enc_bv = (const float*)d_in[14];
  const float* enc_bp = (const float*)d_in[15];
  const float* dec_nq_g = (const float*)d_in[16];
  const float* dec_nq_b = (const float*)d_in[17];
  const float* dec_nk_g = (const float*)d_in[18];
  const float* dec_nk_b = (const float*)d_in[19];
  const float* dec_nv_g = (const float*)d_in[20];
  const float* dec_nv_b = (const float*)d_in[21];
  const float* dec_wq = (const float*)d_in[22];
  const float* dec_wk = (const float*)d_in[23];
  const float* dec_wv = (const float*)d_in[24];
  const float* dec_wp = (const float*)d_in[25];
  const float* dec_bq = (const float*)d_in[26];
  const float* dec_bk = (const float*)d_in[27];
  const float* dec_bv = (const float*)d_in[28];
  const float* dec_bp = (const float*)d_in[29];
  const float* ffn_ng = (const float*)d_in[30];
  const float* ffn_nb = (const float*)d_in[31];
  const float* ffn_w1 = (const float*)d_in[32];
  const float* ffn_b1 = (const float*)d_in[33];
  const float* ffn_w2 = (const float*)d_in[34];
  const float* ffn_b2 = (const float*)d_in[35];
  const float* sr_w1 = (const float*)d_in[36];
  const float* sr_b1 = (const float*)d_in[37];
  const float* sr_w2 = (const float*)d_in[38];
  const float* sr_b2 = (const float*)d_in[39];

  constexpr long S0 = (long)B_ * P_ * C_;   // 25,165,824
  constexpr long SP = (long)B_ * NP_ * C_;  // 786,432
  constexpr long WE = (long)C_ * C_;        // 589,824
  constexpr long WF = (long)C_ * C4_;       // 2,359,296
  const float kScale = 0.03608439182435161f;  // 768^-0.5

  char* ws = (char*)d_ws;
  size_t off = 0;
  auto alloc = [&](size_t bytes) { size_t o = off; off += (bytes + 255) & ~(size_t)255; return o; };
  const size_t o_wbf    = alloc((size_t)(8 * WE + 2 * WF) * 2);  // weights bf16, TRANSPOSED [N][K]
  const size_t o_xt     = alloc((size_t)S0 * 4);
  const size_t o_b1     = alloc((size_t)S0 * 2);  // LN_k(x) -> later qn_dec
  const size_t o_b2     = alloc((size_t)S0 * 2);  // LN_v(x) -> later ctx_dec
  const size_t o_b3     = alloc((size_t)S0 * 2);  // LN_f(x) -> later LN(feats1)
  const size_t o_kn     = alloc((size_t)S0 * 2);  // kn_enc; later feats1 fp32 spans kn+vn
  const size_t o_vn     = alloc((size_t)S0 * 2);  // vn_enc stored TRANSPOSED [B][C][P]
  const size_t o_h1     = alloc((size_t)2048 * C4_ * 2);       // FFN chunk
  const size_t o_S      = alloc((size_t)B_ * NP_ * P_ * 4);    // logits (enc/dec reuse)
  const size_t o_pr     = alloc((size_t)B_ * NP_ * P_ * 2);    // probs bf16 (reuse)
  const size_t o_parts1 = alloc((size_t)SP * 4);
  const size_t o_pin    = alloc((size_t)SP * 4);
  const size_t o_lnp    = alloc((size_t)SP * 2);
  const size_t o_qne    = alloc((size_t)SP * 2);
  const size_t o_ctx    = alloc((size_t)SP * 2);
  const size_t o_dk     = alloc((size_t)SP * 2);
  const size_t o_dv     = alloc((size_t)SP * 2);
  const size_t o_knd    = alloc((size_t)SP * 2);
  const size_t o_vnd    = alloc((size_t)SP * 2);  // vn_dec stored TRANSPOSED [B][C][NP]

  auto F32 = [&](size_t o) { return (float*)(ws + o); };
  auto BF  = [&](size_t o) { return (__bf16*)(ws + o); };
  __bf16* wbf = BF(o_wbf);

  // ---- weights: convert + transpose to bf16 [N][K] ----
  {
    const float* src[10] = {enc_wq, enc_wk, enc_wv, enc_wp, dec_wq, dec_wk, dec_wv, dec_wp,
                            ffn_w1, ffn_w2};
    long offs[10] = {0, WE, 2 * WE, 3 * WE, 4 * WE, 5 * WE, 6 * WE, 7 * WE, 8 * WE, 8 * WE + WF};
    int  Ks[10]   = {C_, C_, C_, C_, C_, C_, C_, C_, C_, C4_};
    int  Ns[10]   = {C_, C_, C_, C_, C_, C_, C_, C_, C4_, C_};
    for (int i = 0; i < 10; ++i)
      cvtT_kernel<<<dim3(Ns[i] / 32, Ks[i] / 32, 1), dim3(32, 8), 0, stream>>>(
          src[i], wbf + offs[i], Ks[i], Ns[i]);
  }
  const __bf16 *w_ewq = wbf, *w_ewk = wbf + WE, *w_ewv = wbf + 2 * WE, *w_ewp = wbf + 3 * WE;
  const __bf16 *w_dwq = wbf + 4 * WE, *w_dwk = wbf + 5 * WE, *w_dwv = wbf + 6 * WE, *w_dwp = wbf + 7 * WE;
  const __bf16 *w_f1 = wbf + 8 * WE, *w_f2 = wbf + 8 * WE + WF;

  // ---- transpose x, LN(x) with three affines ----
  transpose_kernel<<<dim3(P_ / 32, C_ / 32, B_), dim3(32, 8), 0, stream>>>(x, F32(o_xt));
  ln3_kernel<<<B_ * P_, 256, 0, stream>>>(F32(o_xt),
      BF(o_b1), enc_nk_g, enc_nk_b,
      BF(o_b2), enc_nv_g, enc_nv_b,
      BF(o_b3), ffn_ng, ffn_nb);

  // ---- SimpleReasoning gate -> parts1, LN(parts1) ----
  sr_gate_kernel<<<B_, 64, 0, stream>>>(parts, sr_w1, sr_b1, sr_w2, sr_b2, F32(o_parts1));
  ln3_kernel<<<B_ * NP_, 256, 0, stream>>>(F32(o_parts1),
      BF(o_lnp), enc_nq_g, enc_nq_b, nullptr, nullptr, nullptr, nullptr, nullptr, nullptr);

  // ---- enc projections (weights are [N][K]) ----
  gemm_wmma_kernel<2, 4, 0, 0, false><<<dim3(C_ / 256, (B_ * NP_) / 64, 1), 256, 0, stream>>>(
      BF(o_lnp), 0, C_, w_ewq, 0, C_, enc_bq, nullptr, 0, 0, BF(o_qne), 0, C_, C_, 1.f);
  gemm_wmma_kernel<2, 4, 0, 0, false><<<dim3(C_ / 256, (B_ * P_) / 64, 1), 256, 0, stream>>>(
      BF(o_b1), 0, C_, w_ewk, 0, C_, enc_bk, nullptr, 0, 0, BF(o_kn), 0, C_, C_, 1.f);
  // vn written TRANSPOSED [B][C][P] so the ctx GEMM can consume it as [N][K]
  gemm_wmma_kernel<2, 4, 0, 3, false><<<dim3(C_ / 256, (B_ * P_) / 64, 1), 256, 0, stream>>>(
      BF(o_b2), 0, C_, w_ewv, 0, C_, enc_bv, nullptr, 0, 0, BF(o_vn), (long)C_ * P_, P_, C_, 1.f);

  // ---- enc logits S = relu(qn @ kn^T)*scale, softmax (rowmax -> attn_0), ctx ----
  gemm_wmma_kernel<2, 4, 2, 1, false><<<dim3(P_ / 256, 1, B_), 256, 0, stream>>>(
      BF(o_qne), (long)NP_ * C_, C_, BF(o_kn), (long)P_ * C_, C_, nullptr, nullptr, 0, 0,
      F32(o_S), (long)NP_ * P_, P_, C_, kScale);
  softmax_enc_kernel<<<B_ * NP_, 256, 0, stream>>>(F32(o_S), BF(o_pr), (float*)d_out + S0);
  gemm_wmma_kernel<2, 4, 0, 0, false><<<dim3(C_ / 256, 1, B_), 256, 0, stream>>>(
      BF(o_pr), (long)NP_ * P_, P_, BF(o_vn), (long)C_ * P_, P_, nullptr, nullptr, 0, 0,
      BF(o_ctx), (long)NP_ * C_, C_, P_, 1.f);

  // ---- parts_in = parts1 + ctx @ wp + bp ; LN(parts_in) x2 ; dec k/v projections ----
  gemm_wmma_kernel<2, 4, 0, 1, true><<<dim3(C_ / 256, (B_ * NP_) / 64, 1), 256, 0, stream>>>(
      BF(o_ctx), 0, C_, w_ewp, 0, C_, enc_bp, F32(o_parts1), 0, C_,
      F32(o_pin), 0, C_, C_, 1.f);
  ln3_kernel<<<B_ * NP_, 256, 0, stream>>>(F32(o_pin),
      BF(o_dk), dec_nk_g, dec_nk_b, BF(o_dv), dec_nv_g, dec_nv_b, nullptr, nullptr, nullptr);
  gemm_wmma_kernel<2, 4, 0, 0, false><<<dim3(C_ / 256, (B_ * NP_) / 64, 1), 256, 0, stream>>>(
      BF(o_dk), 0, C_, w_dwk, 0, C_, dec_bk, nullptr, 0, 0, BF(o_knd), 0, C_, C_, 1.f);
  // vn_dec written TRANSPOSED [B][C][NP]
  gemm_wmma_kernel<2, 4, 0, 3, false><<<dim3(C_ / 256, (B_ * NP_) / 64, 1), 256, 0, stream>>>(
      BF(o_dv), 0, C_, w_dwv, 0, C_, dec_bv, nullptr, 0, 0, BF(o_vnd), (long)C_ * NP_, NP_, C_, 1.f);

  // ---- FFN (chunked over rows; feats1 fp32 overlays the kn|vn region) ----
  float* feats1 = F32(o_kn);
  for (int ch = 0; ch < 16; ++ch) {
    long r0 = (long)ch * 2048;
    gemm_wmma_kernel<2, 4, 1, 0, false><<<dim3(C4_ / 256, 2048 / 64, 1), 256, 0, stream>>>(
        BF(o_b3) + r0 * C_, 0, C_, w_f1, 0, C_, ffn_b1, nullptr, 0, 0,
        BF(o_h1), 0, C4_, C_, 1.f);
    gemm_wmma_kernel<2, 4, 0, 1, true><<<dim3(C_ / 256, 2048 / 64, 1), 256, 0, stream>>>(
        BF(o_h1), 0, C4_, w_f2, 0, C4_, ffn_b2, F32(o_xt) + r0 * C_, 0, C_,
        feats1 + r0 * C_, 0, C_, C4_, 1.f);
  }

  // ---- dec q projection ----
  ln3_kernel<<<B_ * P_, 256, 0, stream>>>(feats1,
      BF(o_b3), dec_nq_g, dec_nq_b, nullptr, nullptr, nullptr, nullptr, nullptr, nullptr);
  gemm_wmma_kernel<2, 4, 0, 0, false><<<dim3(C_ / 256, (B_ * P_) / 64, 1), 256, 0, stream>>>(
      BF(o_b3), 0, C_, w_dwq, 0, C_, dec_bq, nullptr, 0, 0, BF(o_b1), 0, C_, C_, 1.f);

  // ---- dec logits S2[b,p,np] = relu(qd @ kd^T)*scale, softmax over np, ctx2 ----
  gemm_wmma_kernel<8, 1, 2, 1, false><<<dim3(1, P_ / 256, B_), 256, 0, stream>>>(
      BF(o_b1), (long)P_ * C_, C_, BF(o_knd), (long)NP_ * C_, C_, nullptr, nullptr, 0, 0,
      F32(o_S), (long)P_ * NP_, NP_, C_, kScale);
  softmax_dec_kernel<<<B_ * P_, 64, 0, stream>>>(F32(o_S), BF(o_pr));
  gemm_wmma_kernel<2, 4, 0, 0, false><<<dim3(C_ / 256, P_ / 64, B_), 256, 0, stream>>>(
      BF(o_pr), (long)P_ * NP_, NP_, BF(o_vnd), (long)C_ * NP_, NP_, nullptr, nullptr, 0, 0,
      BF(o_b2), (long)P_ * C_, C_, NP_, 1.f);

  // ---- final: feats = xt + ctx2 @ dwp + dbp, written transposed into d_out[B,C,P] ----
  gemm_wmma_kernel<2, 4, 0, 2, true><<<dim3(C_ / 256, (B_ * P_) / 64, 1), 256, 0, stream>>>(
      BF(o_b2), 0, C_, w_dwp, 0, C_, dec_bp, F32(o_xt), 0, C_,
      (float*)d_out, (long)C_ * P_, P_, C_, 1.f);
}
